// CTBiSSMPolicy_71219147702396
// MI455X (gfx1250) — matrix operand order repaired and checked by reading, not compiled, not run on hardware
//
#include <hip/hip_runtime.h>
#include <hip/hip_bf16.h>
#include <cstdint>

// ---------------------------------------------------------------------------
// CT-SSM policy forward for MI455X (gfx1250).
// GEMMs: v_wmma_f32_16x16x32_bf16. Weight slice staged in LDS via the Tensor
// Data Mover (TDM). Per K-step, all NFRAG B fragments are loaded from LDS in
// one pipelined burst (sched_group_barrier pins DS-reads before WMMAs) so
// only the first fragment pays LDS latency. A double-buffered from global.
// Scan: 3-pass chunked parallel linear recurrence.
// ---------------------------------------------------------------------------

typedef __attribute__((ext_vector_type(16))) __bf16 v16bf;
typedef __attribute__((ext_vector_type(8)))  float  v8f;
typedef __attribute__((ext_vector_type(4)))  unsigned v4u;
typedef __attribute__((ext_vector_type(8)))  int      v8i;
typedef __attribute__((ext_vector_type(4)))  int      v4i;

struct alignas(16) U4 { unsigned x, y, z, w; };
union ABCast { U4 q[2]; v16bf v; };

__device__ __forceinline__ unsigned short f2bf(float f) {
  unsigned u = __builtin_bit_cast(unsigned, f);
  u += 0x7fffu + ((u >> 16) & 1u);              // round-to-nearest-even
  return (unsigned short)(u >> 16);
}

__device__ __forceinline__ float gelu_exact(float x) {
  return 0.5f * x * (1.0f + erff(x * 0.70710678118654752440f));
}

// ---------------------------------------------------------------------------
// WMMA GEMM:  out[M, N] = act(A[M, K] @ W + bias)
//   A  : bf16 row-major, lda = nkt*32
//   Wp : packed bf16, kt-major fragment layout (see pack_w)
//   Block = 256 thr = 8 waves; wave tile = 32 rows x NFRAG*16 cols.
//   gridDim.y splits N into NFRAG-fragment slices.
// ---------------------------------------------------------------------------
template<int NFRAG, int ACT>   // ACT: 0=none 1=gelu 2=sigmoid
__global__ __launch_bounds__(256) void gemm_bf16_wmma(
    const unsigned short* __restrict__ A,
    const unsigned short* __restrict__ Wp,
    const float* __restrict__ bias,
    float* __restrict__ outF,
    unsigned short* __restrict__ outB,
    int ldOutB, int offOutB, int nkt, int nftot)
{
  extern __shared__ char smem[];               // nkt * NFRAG * 1024 bytes
  const int lane    = threadIdx.x & 31;
  const int wave    = threadIdx.x >> 5;
  const int rowbase = blockIdx.x * 256 + wave * 32;
  const int f0      = blockIdx.y * NFRAG;      // fragment slice for this block
  const int coloff  = f0 * 16;
  const int lda     = nkt * 32;

  // ---- stage packed W slice (kt-major, strided 2D tile) into LDS ----
#if __has_builtin(__builtin_amdgcn_tensor_load_to_lds) && __has_builtin(__builtin_amdgcn_s_wait_tensorcnt)
  if (threadIdx.x < 32) {                      // one wave issues the TDM op
    const unsigned long long ga =
        (unsigned long long)(const void*)Wp + (unsigned long long)f0 * 1024ull;
    const unsigned ldsAddr = __builtin_amdgcn_groupstaticsize(); // dyn-LDS base
    const unsigned tdim0 = (unsigned)(nftot * 128);  // tensor row length (qwords)
    const unsigned tdim1 = (unsigned)nkt;
    const unsigned til0  = (unsigned)(NFRAG * 128);  // tile row length (qwords)
    const unsigned til1  = (unsigned)nkt;
    // D# group0: count=1 | lds_addr | global_addr(56:0) | type=2
    v4u g0 = { 1u,
               ldsAddr,
               (unsigned)(ga & 0xFFFFFFFFu),
               (unsigned)((ga >> 32) & 0x01FFFFFFu) | (2u << 30) };
    // D# group1: data_size=3(8B); tensor_dim0[79:48]; tensor_dim1[111:80];
    // tile_dim0[127:112]; tile_dim1[143:128]; tensor_dim0_stride[207:160]
    v8i g1 = { (int)(3u << 16),
               (int)((tdim0 & 0xFFFFu) << 16),
               (int)(((tdim0 >> 16) & 0xFFFFu) | ((tdim1 & 0xFFFFu) << 16)),
               (int)(((tdim1 >> 16) & 0xFFFFu) | (til0 << 16)),
               (int)(til1 & 0xFFFFu),
               (int)tdim0,                     // stride low32 (qwords)
               0, 0 };
    const v4i gz4 = { 0, 0, 0, 0 };
    const v8i gz8 = { 0, 0, 0, 0, 0, 0, 0, 0 };
    __builtin_amdgcn_tensor_load_to_lds(g0, g1, gz4, gz4, gz8, 0);
    __builtin_amdgcn_s_wait_tensorcnt((unsigned short)0);
  }
#else
  {
    const U4* gsrc = (const U4*)Wp;
    U4*       ldst = (U4*)smem;
    const int total4 = nkt * NFRAG * 64;
    for (int i = threadIdx.x; i < total4; i += 256) {
      const int kt  = i / (NFRAG * 64);
      const int rem = i - kt * (NFRAG * 64);
      ldst[i] = gsrc[(size_t)kt * nftot * 64 + (size_t)f0 * 64 + rem];
    }
  }
#endif
  __syncthreads();

  v8f acc[2][NFRAG];
  const v8f vzero = {0.f, 0.f, 0.f, 0.f, 0.f, 0.f, 0.f, 0.f};
#pragma unroll
  for (int rf = 0; rf < 2; ++rf)
#pragma unroll
    for (int f = 0; f < NFRAG; ++f) acc[rf][f] = vzero;

  // A fragment (16x32 bf16): lanes 0-15 rows M=0..15 hold K 0-7 & 16-23,
  // lanes 16-31 same rows hold K 8-15 & 24-31  -> two contiguous b128 loads.
  const unsigned short* aRow0 =
      A + (size_t)(rowbase + (lane & 15)) * (size_t)lda + ((lane >> 4) << 3);
  const unsigned short* aRow1 = aRow0 + (size_t)16 * lda;

  ABCast aCur[2], aNxt[2];
  aCur[0].q[0] = *(const U4*)(aRow0);
  aCur[0].q[1] = *(const U4*)(aRow0 + 16);
  aCur[1].q[0] = *(const U4*)(aRow1);
  aCur[1].q[1] = *(const U4*)(aRow1 + 16);

  const char* bBase = smem + lane * 32;        // per-lane 32B B-fragment chunk

  for (int kt = 0; kt < nkt; ++kt) {
    if (kt + 1 < nkt) {                        // prefetch next A (global)
      aNxt[0].q[0] = *(const U4*)(aRow0 + (kt + 1) * 32);
      aNxt[0].q[1] = *(const U4*)(aRow0 + (kt + 1) * 32 + 16);
      aNxt[1].q[0] = *(const U4*)(aRow1 + (kt + 1) * 32);
      aNxt[1].q[1] = *(const U4*)(aRow1 + (kt + 1) * 32 + 16);
    }
    const char* bkt = bBase + kt * (NFRAG * 1024);
    // load ALL B fragments of this K-step into live registers so the
    // ds_loads pipeline and WMMA waits are staggered, not back-to-back
    ABCast bf[NFRAG];
#pragma unroll
    for (int f = 0; f < NFRAG; ++f) {
      bf[f].q[0] = *(const U4*)(bkt + f * 1024);
      bf[f].q[1] = *(const U4*)(bkt + f * 1024 + 16);
    }
#pragma unroll
    for (int f = 0; f < NFRAG; ++f) {
      acc[0][f] = __builtin_amdgcn_wmma_f32_16x16x32_bf16(
          false, aCur[0].v, false, bf[f].v, (short)0, acc[0][f], false, false);
      acc[1][f] = __builtin_amdgcn_wmma_f32_16x16x32_bf16(
          false, aCur[1].v, false, bf[f].v, (short)0, acc[1][f], false, false);
    }
#if __has_builtin(__builtin_amdgcn_sched_group_barrier)
    // Pin the pipeline: all DS reads of this K-step first, then the WMMAs.
    // 0x100 = DS read, 0x008 = MFMA/WMMA.
    __builtin_amdgcn_sched_group_barrier(0x100, 2 * NFRAG, 0);
    __builtin_amdgcn_sched_group_barrier(0x008, 2 * NFRAG, 0);
#endif
    if (kt + 1 < nkt) { aCur[0] = aNxt[0]; aCur[1] = aNxt[1]; }
  }

  // C/D layout: lanes 0-15: M = r (VGPR r), N = lane; lanes 16-31: M = 8+r.
  const int rsub = ((lane >> 4) << 3);
  const int col0 = lane & 15;
#pragma unroll
  for (int rf = 0; rf < 2; ++rf) {
    const int rowb = rowbase + rf * 16 + rsub;
#pragma unroll
    for (int f = 0; f < NFRAG; ++f) {
      const int col = coloff + f * 16 + col0;
      const float bc = bias[col];
#pragma unroll
      for (int r = 0; r < 8; ++r) {
        float v = acc[rf][f][r] + bc;
        if (ACT == 1)      v = gelu_exact(v);
        else if (ACT == 2) v = 1.0f / (1.0f + expf(-v));
        const int row = rowb + r;
        if (outF) outF[(size_t)row * (nftot * 16) + col] = v;
        if (outB) outB[(size_t)row * ldOutB + offOutB + col] = f2bf(v);
      }
    }
  }
}

// Pack fp32 weight W[K,N] (row-major) -> bf16 fragment layout, kt-major:
// ushort offset ((kt*nfrag + f)*32 + lane)*16 + 2j  holds (W[k][n], W[k+1][n])
// with n = f*16 + (lane&15), k = kt*32 + (lane>>4)*8 + 2j + (j>>2)*8. k >= K -> zero.
__global__ void pack_w(const float* __restrict__ W, unsigned short* __restrict__ out,
                       int K, int N, int nkt, int nfrag)
{
  const int tid = blockIdx.x * blockDim.x + threadIdx.x;
  if (tid >= nfrag * nkt * 256) return;
  const int j    = tid & 7;
  const int lane = (tid >> 3) & 31;
  const int t    = tid >> 8;
  const int kt   = t % nkt;
  const int f    = t / nkt;
  const int n    = f * 16 + (lane & 15);
  const int k    = kt * 32 + ((lane >> 4) << 3) + 2 * j + ((j >> 2) << 3);
  const unsigned short lo = (k     < K) ? f2bf(W[(size_t)k * N + n])       : (unsigned short)0;
  const unsigned short hi = (k + 1 < K) ? f2bf(W[(size_t)(k + 1) * N + n]) : (unsigned short)0;
  const size_t o = (((size_t)(kt * nfrag + f) * 32) + lane) * 16 + 2 * j;
  out[o] = lo; out[o + 1] = hi;
}

// fp32 [M,kin] -> bf16 [M,kpad] with zero padding
__global__ void cvt_f32_bf16_pad(const float* __restrict__ in, unsigned short* __restrict__ out,
                                 int kin, int kpad, int total)
{
  const int i = blockIdx.x * blockDim.x + threadIdx.x;
  if (i >= total) return;
  const int m = i / kpad, c = i - m * kpad;
  out[i] = (c < kin) ? f2bf(in[(size_t)m * kin + c]) : (unsigned short)0;
}

// out[m, off+j] = bf16(x[m]*w[j] + b[j])  (Linear(1, dim), writes into concat)
__global__ void scalar_linear_bf16(const float* __restrict__ x, const float* __restrict__ w,
                                   const float* __restrict__ b, unsigned short* __restrict__ out,
                                   int ld, int off, int dim, int total)
{
  const int i = blockIdx.x * blockDim.x + threadIdx.x;
  if (i >= total) return;
  const int m = i / dim, j = i - m * dim;
  out[(size_t)m * ld + off + j] = f2bf(x[m] * w[j] + b[j]);
}

__global__ void zero_bf16_cols(unsigned short* __restrict__ out, int ld, int off,
                               int width, int total)
{
  const int i = blockIdx.x * blockDim.x + threadIdx.x;
  if (i >= total) return;
  const int m = i / width, j = i - m * width;
  out[(size_t)m * ld + off + j] = 0;
}

// ---------------------------------------------------------------------------
// Chunked parallel scan of s_t = lamb_t * s_{t-1} + drive_t over T=1024.
// ---------------------------------------------------------------------------
#define SCAN_T   1024
#define SCAN_NCH 16
#define SCAN_CK  64
#define SCAN_D   256

__device__ __forceinline__ float decay_of(float x) {
  float d = (x > 20.f) ? x : log1pf(expf(x));
  return d + 1e-4f;
}

__global__ __launch_bounds__(256) void ssm_scan_pass1(
    const float* __restrict__ upd, const float* __restrict__ gate,
    const float* __restrict__ dt,  const float* __restrict__ log_decay,
    float* __restrict__ localF, float* __restrict__ Pc, float* __restrict__ lastc)
{
  const int b  = blockIdx.x / SCAN_NCH;
  const int ch = blockIdx.x - b * SCAN_NCH;
  const int d  = threadIdx.x;
  const float decay = decay_of(log_decay[d]);
  const size_t tb = (size_t)b * SCAN_T + (size_t)ch * SCAN_CK;
  size_t idx = tb * SCAN_D + d;
  float s = 0.f, P = 1.f;
  for (int t = 0; t < SCAN_CK; ++t, idx += SCAN_D) {
    const float dd = fmaxf(dt[tb + t], 1e-4f);
    const float l  = expf(-decay * dd);
    s = l * s + (1.f - l) * gate[idx] * upd[idx];
    P *= l;
    localF[idx] = s;
  }
  const size_t cs = ((size_t)b * SCAN_NCH + ch) * SCAN_D + d;
  Pc[cs] = P;
  lastc[cs] = s;
}

__global__ __launch_bounds__(256) void ssm_scan_pass2(
    const float* __restrict__ Pc, const float* __restrict__ lastc,
    float* __restrict__ carry)
{
  const int b = blockIdx.x;
  const int d = threadIdx.x;
  float c = 0.f;
  carry[(size_t)b * SCAN_NCH * SCAN_D + d] = 0.f;
  for (int ch = 1; ch < SCAN_NCH; ++ch) {
    const size_t p = ((size_t)b * SCAN_NCH + ch - 1) * SCAN_D + d;
    c = lastc[p] + Pc[p] * c;
    carry[((size_t)b * SCAN_NCH + ch) * SCAN_D + d] = c;
  }
}

__global__ __launch_bounds__(256) void ssm_scan_pass3(
    const float* __restrict__ dt, const float* __restrict__ log_decay,
    const float* __restrict__ carry,
    float* __restrict__ sF, unsigned short* __restrict__ sB)
{
  const int b  = blockIdx.x / SCAN_NCH;
  const int ch = blockIdx.x - b * SCAN_NCH;
  const int d  = threadIdx.x;
  const float decay = decay_of(log_decay[d]);
  const float cin = carry[((size_t)b * SCAN_NCH + ch) * SCAN_D + d];
  const size_t tb = (size_t)b * SCAN_T + (size_t)ch * SCAN_CK;
  size_t idx = tb * SCAN_D + d;
  float P = 1.f;
  for (int t = 0; t < SCAN_CK; ++t, idx += SCAN_D) {
    const float dd = fmaxf(dt[tb + t], 1e-4f);
    P *= expf(-decay * dd);
    const float s = sF[idx] + P * cin;
    sF[idx] = s;
    sB[idx] = f2bf(s);
  }
}

// LayerNorm((x + y)) over 256 channels; one wave32 per row, shfl reductions.
__global__ __launch_bounds__(256) void add_layernorm(
    const float* __restrict__ x, const float* __restrict__ y,
    const float* __restrict__ g, const float* __restrict__ b,
    float* __restrict__ outF, unsigned short* __restrict__ outB,
    int ldOutB, int offOutB)
{
  const int lane = threadIdx.x & 31;
  const int wave = threadIdx.x >> 5;
  const size_t row  = (size_t)blockIdx.x * 8 + wave;
  const size_t base = row * 256;
  float v[8];
  float s = 0.f, s2 = 0.f;
#pragma unroll
  for (int i = 0; i < 8; ++i) {
    const int c = lane + i * 32;
    float t = x[base + c];
    if (y) t += y[base + c];
    v[i] = t; s += t; s2 += t * t;
  }
#pragma unroll
  for (int m = 16; m >= 1; m >>= 1) {
    s  += __shfl_xor(s,  m, 32);
    s2 += __shfl_xor(s2, m, 32);
  }
  const float mu  = s * (1.f / 256.f);
  const float var = s2 * (1.f / 256.f) - mu * mu;
  const float inv = rsqrtf(var + 1e-5f);
#pragma unroll
  for (int i = 0; i < 8; ++i) {
    const int c = lane + i * 32;
    const float o = (v[i] - mu) * inv * g[c] + b[c];
    if (outF) outF[base + c] = o;
    if (outB) outB[row * ldOutB + offOutB + c] = f2bf(o);
  }
}

// Split head output [M,32] -> mean [M,16], clipped log_std [M,16]
__global__ void head_split(const float* __restrict__ h, float* __restrict__ mean,
                           float* __restrict__ logstd, int total)
{
  const int i = blockIdx.x * blockDim.x + threadIdx.x;
  if (i >= total) return;
  const int m = i >> 5, c = i & 31;
  float v = h[i];
  if (c < 16) mean[m * 16 + c] = v;
  else        logstd[m * 16 + (c - 16)] = fminf(fmaxf(v, -5.f), 2.f);
}

// ---------------------------------------------------------------------------
extern "C" void kernel_launch(void* const* d_in, const int* in_sizes, int n_in,
                              void* d_out, int out_size, void* d_ws, size_t ws_size,
                              hipStream_t stream)
{
  (void)in_sizes; (void)n_in; (void)out_size; (void)ws_size;
  constexpr int    Mi = 32 * 1024;       // B*T rows
  constexpr size_t M  = (size_t)Mi;
  constexpr int    GB = Mi / 256;        // GEMM grid.x (256 rows/block)

  auto F = [&](int i) -> const float* { return (const float*)d_in[i]; };

  // positional tensors (setup_inputs dict order)
  const float* states  = F(0);
  const float* actions = F(1);
  const float* rewards = F(2);
  const float* rtg     = F(3);
  const float* logdt   = F(4);
  const float* dones   = F(5);
  const float* dt      = F(6);

  // params leaves (jax tree order: sorted dict keys, lists in order, {'b','w'}, {'b','g'})
  const float *ac0_b = F(7),  *ac0_w = F(8),  *ac1_b = F(9),  *ac1_w = F(10);
  const float *dn_b  = F(11), *dn_w  = F(12);
  const float *fln_b = F(13), *fln_g = F(14);
  const float *h0_b  = F(15), *h0_w  = F(16);
  const float *h1_b  = F(17), *h1_w  = F(18);
  const float *h2_b  = F(19), *h2_w  = F(20);
  // layers: base 21 + 9*L -> gate.b, gate.w, inp.b, inp.w, ln.b, ln.g, log_decay, out.b, out.w
  const float *proj_b = F(57), *proj_w = F(58);
  const float *rtg_b  = F(59), *rtg_w  = F(60);
  const float *rtg2_b = F(61), *rtg2_w = F(62);
  const float *rw_b   = F(63), *rw_w   = F(64);
  const float *st0_b  = F(65), *st0_w  = F(66);
  const float *st1_b  = F(67), *st1_w  = F(68);
  const float *tm_b   = F(69), *tm_w   = F(70);

  // --- workspace bump allocator (256B aligned) ---
  char* ws = (char*)d_ws;
  size_t off = 0;
  auto alloc = [&](size_t bytes) -> char* {
    size_t o = (off + 255) & ~(size_t)255;
    off = o + bytes;
    return ws + o;
  };
  typedef unsigned short u16;

  // packed weights (bf16, kt-major fragment layout)
  u16* pk_st0  = (u16*)alloc((size_t)64  * 256 * 2);
  u16* pk_st1  = (u16*)alloc((size_t)256 * 256 * 2);
  u16* pk_ac0  = (u16*)alloc((size_t)32  * 128 * 2);
  u16* pk_ac1  = (u16*)alloc((size_t)128 * 128 * 2);
  u16* pk_proj = (u16*)alloc((size_t)608 * 256 * 2);
  u16 *pk_gate[4], *pk_inp[4], *pk_out[4];
  for (int L = 0; L < 4; ++L) {
    pk_gate[L] = (u16*)alloc((size_t)256 * 256 * 2);
    pk_inp[L]  = (u16*)alloc((size_t)256 * 256 * 2);
    pk_out[L]  = (u16*)alloc((size_t)256 * 256 * 2);
  }
  u16* pk_h0 = (u16*)alloc((size_t)320 * 256 * 2);
  u16* pk_h1 = (u16*)alloc((size_t)256 * 256 * 2);
  u16* pk_h2 = (u16*)alloc((size_t)256 * 32  * 2);

  // activations
  u16*   stIn   = (u16*)alloc(M * 64  * 2);
  u16*   acIn   = (u16*)alloc(M * 32  * 2);
  u16*   st1B   = (u16*)alloc(M * 256 * 2);
  u16*   ac1B   = (u16*)alloc(M * 128 * 2);
  u16*   feat   = (u16*)alloc(M * 608 * 2);   // [st|ac|rw|rtg|tm|dn|pad16]
  u16*   hbf    = (u16*)alloc(M * 256 * 2);   // tokens / LN outputs (bf16)
  float* hF     = (float*)alloc(M * 256 * 4); // last block LN output (fp32)
  float* updF   = (float*)alloc(M * 256 * 4);
  float* gateF  = (float*)alloc(M * 256 * 4);
  float* stF    = (float*)alloc(M * 256 * 4);
  u16*   stB    = (u16*)alloc(M * 256 * 2);
  float* yF     = (float*)alloc(M * 256 * 4);
  u16*   headin = (u16*)alloc(M * 320 * 2);   // [hidden(256)|rtg2(64)]
  u16*   headA  = (u16*)alloc(M * 256 * 2);
  u16*   headB  = (u16*)alloc(M * 256 * 2);
  float* h3F    = (float*)alloc(M * 32  * 4);
  // scan chunk summaries: [B, NCH, D]
  float* scanP  = (float*)alloc((size_t)32 * 16 * 256 * 4);
  float* scanL  = (float*)alloc((size_t)32 * 16 * 256 * 4);
  float* scanC  = (float*)alloc((size_t)32 * 16 * 256 * 4);

  float* out_hidden = (float*)d_out;
  float* out_mean   = out_hidden + M * 256;
  float* out_logstd = out_mean   + M * 16;

  // --- pack all weights to bf16 fragment layout ---
  auto pack = [&](const float* W, u16* dst, int K, int N, int nkt) {
    const int nfrag = N / 16;
    pack_w<<<dim3(nfrag * nkt), dim3(256), 0, stream>>>(W, dst, K, N, nkt, nfrag);
  };
  pack(st0_w,  pk_st0,  64,  256, 2);
  pack(st1_w,  pk_st1,  256, 256, 8);
  pack(ac0_w,  pk_ac0,  16,  128, 1);
  pack(ac1_w,  pk_ac1,  128, 128, 4);
  pack(proj_w, pk_proj, 592, 256, 19);
  for (int L = 0; L < 4; ++L) {
    const int base = 21 + 9 * L;
    pack(F(base + 1), pk_gate[L], 256, 256, 8);
    pack(F(base + 3), pk_inp[L],  256, 256, 8);
    pack(F(base + 8), pk_out[L],  256, 256, 8);
  }
  pack(h0_w, pk_h0, 320, 256, 10);
  pack(h1_w, pk_h1, 256, 256, 8);
  pack(h2_w, pk_h2, 256, 32,  8);

  // GEMM launcher: NFRAG tiles of 16 cols, grid.y splits N (N=256 -> 2 slices)
#define GEMM(NF, ACT, Ain, Wpk, Bias, OutF, OutB, ldB, offB, NKT, NFTOT, GY)        \
  gemm_bf16_wmma<NF, ACT><<<dim3(GB, GY), dim3(256), (unsigned)((NKT) * (NF) * 1024), stream>>>( \
      Ain, Wpk, Bias, OutF, OutB, ldB, offB, NKT, NFTOT)

  // --- token encoder ---
  cvt_f32_bf16_pad<<<dim3((Mi * 64 + 255) / 256), dim3(256), 0, stream>>>(states, stIn, 64, 64, Mi * 64);
  cvt_f32_bf16_pad<<<dim3((Mi * 32 + 255) / 256), dim3(256), 0, stream>>>(actions, acIn, 16, 32, Mi * 32);

  GEMM(8, 1, stIn, pk_st0, st0_b, nullptr, st1B, 256, 0, 2, 16, 2);
  GEMM(8, 0, st1B, pk_st1, st1_b, nullptr, feat, 608, 0, 8, 16, 2);
  GEMM(8, 1, acIn, pk_ac0, ac0_b, nullptr, ac1B, 128, 0, 1, 8, 1);
  GEMM(8, 0, ac1B, pk_ac1, ac1_b, nullptr, feat, 608, 256, 4, 8, 1);

  scalar_linear_bf16<<<dim3((Mi * 64 + 255) / 256), dim3(256), 0, stream>>>(rewards, rw_w,  rw_b,  feat, 608, 384, 64, Mi * 64);
  scalar_linear_bf16<<<dim3((Mi * 64 + 255) / 256), dim3(256), 0, stream>>>(rtg,     rtg_w, rtg_b, feat, 608, 448, 64, Mi * 64);
  scalar_linear_bf16<<<dim3((Mi * 64 + 255) / 256), dim3(256), 0, stream>>>(logdt,   tm_w,  tm_b,  feat, 608, 512, 64, Mi * 64);
  scalar_linear_bf16<<<dim3((Mi * 16 + 255) / 256), dim3(256), 0, stream>>>(dones,   dn_w,  dn_b,  feat, 608, 576, 16, Mi * 16);
  zero_bf16_cols<<<dim3((Mi * 16 + 255) / 256), dim3(256), 0, stream>>>(feat, 608, 592, 16, Mi * 16);

  GEMM(8, 0, feat, pk_proj, proj_b, nullptr, hbf, 256, 0, 19, 16, 2);

  // --- CT-SSM backbone (4 layers) ---
  for (int L = 0; L < 4; ++L) {
    const int base = 21 + 9 * L;
    const float* gate_b = F(base + 0);
    const float* inp_b  = F(base + 2);
    const float* ln_b   = F(base + 4);
    const float* ln_g   = F(base + 5);
    const float* ldec   = F(base + 6);
    const float* outw_b = F(base + 7);

    GEMM(8, 0, hbf, pk_inp[L],  inp_b,  updF,  nullptr, 0, 0, 8, 16, 2);
    GEMM(8, 2, hbf, pk_gate[L], gate_b, gateF, nullptr, 0, 0, 8, 16, 2);
    ssm_scan_pass1<<<dim3(32 * SCAN_NCH), dim3(256), 0, stream>>>(updF, gateF, dt, ldec, stF, scanP, scanL);
    ssm_scan_pass2<<<dim3(32), dim3(256), 0, stream>>>(scanP, scanL, scanC);
    ssm_scan_pass3<<<dim3(32 * SCAN_NCH), dim3(256), 0, stream>>>(dt, ldec, scanC, stF, stB);
    GEMM(8, 0, stB, pk_out[L], outw_b, yF, nullptr, 0, 0, 8, 16, 2);
    add_layernorm<<<dim3(Mi / 8), dim3(256), 0, stream>>>(stF, yF, ln_g, ln_b,
                                                          (L == 3) ? hF : nullptr, hbf, 256, 0);
  }

  // --- final LN: hidden -> d_out (fp32) and headin[:,0:256] (bf16) ---
  add_layernorm<<<dim3(Mi / 8), dim3(256), 0, stream>>>(hF, nullptr, fln_g, fln_b,
                                                        out_hidden, headin, 320, 0);
  // rtg2 embedding -> headin[:,256:320]
  scalar_linear_bf16<<<dim3((Mi * 64 + 255) / 256), dim3(256), 0, stream>>>(rtg, rtg2_w, rtg2_b, headin, 320, 256, 64, Mi * 64);

  // --- Gaussian action head ---
  GEMM(8, 1, headin, pk_h0, h0_b, nullptr, headA, 256, 0, 10, 16, 2);
  GEMM(8, 1, headA,  pk_h1, h1_b, nullptr, headB, 256, 0, 8, 16, 2);
  GEMM(2, 0, headB,  pk_h2, h2_b, h3F,    nullptr, 0, 0, 8, 2, 1);

  head_split<<<dim3((Mi * 32 + 255) / 256), dim3(256), 0, stream>>>(h3F, out_mean, out_logstd, Mi * 32);
#undef GEMM
}
